// GroupedQueryAttention_45148696217023
// MI455X (gfx1250) — compile-verified, matrix operand order checked by reading, not run on hardware
//
#include <hip/hip_runtime.h>

// ---------------------------------------------------------------------------
// GQA forward for MI455X (gfx1250, wave32, WMMA).
// All matmuls in bf16 WMMA (v_wmma_f32_16x16x32_bf16) with f32 accumulation.
// GEMM: 32(M) x 64(N) tile per wave -> 8 WMMA per 12 B128 loads per K-step.
// ---------------------------------------------------------------------------

typedef __attribute__((ext_vector_type(16))) __bf16 v16bf;
typedef __attribute__((ext_vector_type(8)))  __bf16 v8bf;
typedef __attribute__((ext_vector_type(4)))  __bf16 v4bf;
typedef __attribute__((ext_vector_type(8)))  float  v8f;

#define B_   2
#define S_   2048
#define D_   4096
#define NH   32
#define NKV  8
#define HD   128
// NH*HD == D_ == 4096, G = NH/NKV = 4

// ---- fragment loader: 16-bit A/B fragment (16 rows x 32 K) ----------------
// Lane L: row = L&15, half = L>>4.  Elements 0..7  = cols col0 + half*8 + e
//                                   Elements 8..15 = cols col0 + 16 + half*8 + e
__device__ __forceinline__ v16bf load_frag16(const __bf16* rowptr, int col0, int half) {
    v8bf lo = *(const v8bf*)(rowptr + col0 + half * 8);
    v8bf hi = *(const v8bf*)(rowptr + col0 + 16 + half * 8);
    return __builtin_shufflevector(lo, hi, 0, 1, 2, 3, 4, 5, 6, 7,
                                           8, 9, 10, 11, 12, 13, 14, 15);
}

__device__ __forceinline__ v8f wmma_bf16(v16bf a, v16bf b, v8f c) {
    // (neg_a, A, neg_b, B, c_mod, C, reuse_a, reuse_b)
    return __builtin_amdgcn_wmma_f32_16x16x32_bf16(false, a, false, b,
                                                   (short)0, c, false, false);
}

// ---- elementwise f32 -> bf16 ----------------------------------------------
__global__ __launch_bounds__(256) void cvt_f32_bf16(const float* __restrict__ in,
                                                    __bf16* __restrict__ out, int n) {
    int i = (blockIdx.x * 256 + threadIdx.x) * 4;
    if (i >= n) return;
    float4 f = *(const float4*)(in + i);
    v4bf o;
    o.x = (__bf16)f.x; o.y = (__bf16)f.y; o.z = (__bf16)f.z; o.w = (__bf16)f.w;
    *(v4bf*)(out + i) = o;
}

// ---- tiled transpose + convert: in[K][N] f32 -> out[N][K] bf16 ------------
__global__ __launch_bounds__(256) void transpose_cvt(const float* __restrict__ in,
                                                     __bf16* __restrict__ out,
                                                     int K, int N) {
    __shared__ float tile[32][33];
    int n0 = blockIdx.x * 32, k0 = blockIdx.y * 32;
    #pragma unroll
    for (int i = 0; i < 4; ++i)
        tile[threadIdx.y + i * 8][threadIdx.x] =
            in[(size_t)(k0 + threadIdx.y + i * 8) * N + n0 + threadIdx.x];
    __syncthreads();
    #pragma unroll
    for (int i = 0; i < 4; ++i)
        out[(size_t)(n0 + threadIdx.y + i * 8) * K + k0 + threadIdx.x] =
            (__bf16)tile[threadIdx.x][threadIdx.y + i * 8];
}

// ---- WMMA GEMM: C[M x N] = A[M x K] * Bt[N x K]^T -------------------------
// One wave computes a 32(M) x 64(N) tile: 2 A-frags x 4 B-frags = 8 WMMA per
// 32-deep K step, 12 B128 loads.  Block = 4 waves (128 threads), each wave
// owns a different 64-wide N chunk.
// MODE 0: plain bf16 out [M x N]           (Q projection -> qb)
// MODE 1: f32 kv-cache layout + bf16 same  (K projection -> k_cache, kb)
// MODE 2: f32 kv-cache layout + bf16 T     (V projection -> v_cache, vt)
// MODE 3: plain f32 out [M x N]            (O projection -> out)
template <int MODE>
__global__ __launch_bounds__(128) void gemm_bf16(const __bf16* __restrict__ A,
                                                 const __bf16* __restrict__ Bt,
                                                 int M, int N, int K,
                                                 float* __restrict__ outF,
                                                 __bf16* __restrict__ outB) {
    int tid = threadIdx.x, lane = tid & 31, wave = tid >> 5;
    int lm = lane & 15, half = lane >> 4;
    int m0 = blockIdx.y * 32;
    int n0 = (blockIdx.x * 4 + wave) * 64;

    const __bf16* arow0 = A + (size_t)(m0 + lm) * K;
    const __bf16* arow1 = A + (size_t)(m0 + 16 + lm) * K;
    const __bf16* brow0 = Bt + (size_t)(n0 + lm) * K;

    v8f acc[2][4] = {};
    for (int kk = 0; kk < K; kk += 32) {
        v16bf a0 = load_frag16(arow0, kk, half);
        v16bf a1 = load_frag16(arow1, kk, half);
        __builtin_prefetch(arow0 + kk + 256, 0, 1);   // global_prefetch_b8
        #pragma unroll
        for (int j = 0; j < 4; ++j) {
            v16bf b = load_frag16(brow0 + (size_t)j * 16 * K, kk, half);
            acc[0][j] = wmma_bf16(a0, b, acc[0][j]);
            acc[1][j] = wmma_bf16(a1, b, acc[1][j]);
        }
    }

    // C/D layout: lane holds col c = n0+j*16+lm, rows m = i*16 + half*8 + v
    #pragma unroll
    for (int i = 0; i < 2; ++i) {
        #pragma unroll
        for (int j = 0; j < 4; ++j) {
            #pragma unroll
            for (int v = 0; v < 8; ++v) {
                int r = m0 + i * 16 + half * 8 + v;
                int c = n0 + j * 16 + lm;
                float val = acc[i][j][v];
                if (MODE == 0) {
                    outB[(size_t)r * N + c] = (__bf16)val;
                } else if (MODE == 3) {
                    outF[(size_t)r * N + c] = val;
                } else {
                    int bb = r / S_, srow = r % S_;
                    int h = c >> 7, d = c & 127;
                    size_t ci = (((size_t)(bb * NKV + h)) * S_ + srow) * HD + d;
                    outF[ci] = val;                               // fp32 kv cache
                    if (MODE == 1)
                        outB[ci] = (__bf16)val;                   // kb [b,kv,s,d]
                    else
                        outB[((size_t)(bb * NKV + h) * HD + d) * S_ + srow] =
                            (__bf16)val;                          // vt [b,kv,d,s]
                }
            }
        }
    }
}

// ---- Flash-style causal GQA attention -------------------------------------
// Grid: x = S/16/4 q-tile groups, y = B*NH (batch*head). 1 wave = 16 q rows.
__global__ __launch_bounds__(128) void gqa_flash(const __bf16* __restrict__ qb,
                                                 const __bf16* __restrict__ kbuf,
                                                 const __bf16* __restrict__ vtb,
                                                 __bf16* __restrict__ ab) {
    __shared__ __bf16 ldsP[4][16][40];   // row stride 80B -> 16B aligned chunks
    int tid = threadIdx.x, lane = tid & 31, wave = tid >> 5;
    int lm = lane & 15, half = lane >> 4;
    int q0 = (blockIdx.x * 4 + wave) * 16;
    int b = blockIdx.y >> 5, head = blockIdx.y & 31, kvh = head >> 2;
    const float scale = 0.08838834764831845f;   // 1/sqrt(128)

    // Q fragments: 16 x 128 bf16, held in registers for the whole pass
    const __bf16* qrow = qb + ((size_t)(b * S_ + q0 + lm)) * D_ + head * HD;
    v16bf qf[4];
    #pragma unroll
    for (int kk = 0; kk < 4; ++kk) qf[kk] = load_frag16(qrow, kk * 32, half);

    const __bf16* kbase = kbuf + (size_t)(b * NKV + kvh) * S_ * HD;  // [s][d]
    const __bf16* vbase = vtb + (size_t)(b * NKV + kvh) * HD * S_;   // [d][s]

    v8f acco[8] = {};
    float mrow[8], lrow[8];
    #pragma unroll
    for (int v = 0; v < 8; ++v) { mrow[v] = -1e30f; lrow[v] = 0.f; }

    int nblocks = (q0 + 16 + 31) >> 5;   // causal: key blocks of 32
    for (int blk = 0; blk < nblocks; ++blk) {
        int key0 = blk * 32;

        // scores: S = Q * K^T   (B fragment rows = K rows, contiguous)
        v8f s[2] = {};
        #pragma unroll
        for (int t = 0; t < 2; ++t) {
            const __bf16* krow = kbase + (size_t)(key0 + t * 16 + lm) * HD;
            #pragma unroll
            for (int kk = 0; kk < 4; ++kk) {
                v16bf kf = load_frag16(krow, kk * 32, half);
                s[t] = wmma_bf16(qf[kk], kf, s[t]);
            }
        }

        // scale, causal mask, per-row block max
        float bmax[8];
        #pragma unroll
        for (int v = 0; v < 8; ++v) bmax[v] = -1e30f;
        #pragma unroll
        for (int t = 0; t < 2; ++t) {
            int key = key0 + t * 16 + lm;
            #pragma unroll
            for (int v = 0; v < 8; ++v) {
                int qpos = q0 + half * 8 + v;
                float val = s[t][v] * scale;
                if (key > qpos) val = -1e30f;
                s[t][v] = val;
                bmax[v] = fmaxf(bmax[v], val);
            }
        }
        #pragma unroll
        for (int v = 0; v < 8; ++v)
            #pragma unroll
            for (int msk = 1; msk < 16; msk <<= 1)   // stays inside 16-lane half
                bmax[v] = fmaxf(bmax[v], __shfl_xor(bmax[v], msk, 32));

        // online softmax update
        float corr[8], rsum[8];
        #pragma unroll
        for (int v = 0; v < 8; ++v) {
            float mnew = fmaxf(mrow[v], bmax[v]);
            corr[v] = __expf(mrow[v] - mnew);
            mrow[v] = mnew;
            rsum[v] = 0.f;
        }

        asm volatile("" ::: "memory");   // pin LDS write block below prior reads
        #pragma unroll
        for (int t = 0; t < 2; ++t)
            #pragma unroll
            for (int v = 0; v < 8; ++v) {
                float p = __expf(s[t][v] - mrow[v]);
                rsum[v] += p;
                ldsP[wave][half * 8 + v][t * 16 + lm] = (__bf16)p;
            }
        #pragma unroll
        for (int v = 0; v < 8; ++v) {
            #pragma unroll
            for (int msk = 1; msk < 16; msk <<= 1)
                rsum[v] += __shfl_xor(rsum[v], msk, 32);
            lrow[v] = lrow[v] * corr[v] + rsum[v];
        }
        #pragma unroll
        for (int j = 0; j < 8; ++j)
            #pragma unroll
            for (int v = 0; v < 8; ++v) acco[j][v] *= corr[v];

        // same-wave LDS RAW: DS is in-order per wave; wait + compiler barrier
        asm volatile("s_wait_dscnt 0" ::: "memory");

        // P (C-layout -> A-layout through LDS), then O += P * V
        v8bf plo = *(const v8bf*)&ldsP[wave][lm][half * 8];
        v8bf phi = *(const v8bf*)&ldsP[wave][lm][16 + half * 8];
        v16bf pa = __builtin_shufflevector(plo, phi, 0, 1, 2, 3, 4, 5, 6, 7,
                                                     8, 9, 10, 11, 12, 13, 14, 15);
        #pragma unroll
        for (int j = 0; j < 8; ++j) {
            const __bf16* vrow = vbase + (size_t)(j * 16 + lm) * S_;  // row = d
            v16bf vf = load_frag16(vrow, key0, half);
            acco[j] = wmma_bf16(pa, vf, acco[j]);
        }
    }

    // normalize and write attention output (bf16, [b*s][NH*HD])
    float inv[8];
    #pragma unroll
    for (int v = 0; v < 8; ++v) inv[v] = 1.0f / lrow[v];
    #pragma unroll
    for (int j = 0; j < 8; ++j)
        #pragma unroll
        for (int v = 0; v < 8; ++v)
            ab[(size_t)(b * S_ + q0 + half * 8 + v) * D_ + head * HD + j * 16 + lm] =
                (__bf16)(acco[j][v] * inv[v]);
}

// ---------------------------------------------------------------------------
extern "C" void kernel_launch(void* const* d_in, const int* in_sizes, int n_in,
                              void* d_out, int out_size, void* d_ws, size_t ws_size,
                              hipStream_t stream) {
    const float* x  = (const float*)d_in[0];
    const float* Wq = (const float*)d_in[1];
    const float* Wk = (const float*)d_in[2];
    const float* Wv = (const float*)d_in[3];
    const float* Wo = (const float*)d_in[4];

    float* out     = (float*)d_out;                       // [B,S,D]
    float* k_cache = out + (size_t)B_ * S_ * D_;          // [B,NKV,S,HD]
    float* v_cache = k_cache + (size_t)B_ * NKV * S_ * HD;

    // workspace layout (bf16), ~192 MiB total
    const size_t BIG = (size_t)4096 * 4096;   // 16,777,216
    const size_t SML = (size_t)4096 * 1024;   //  4,194,304
    __bf16* xb  = (__bf16*)d_ws;      // x,  [B*S][D]
    __bf16* wqt = xb  + BIG;          // Wq^T [4096][4096]
    __bf16* wkt = wqt + BIG;          // Wk^T [1024][4096]
    __bf16* wvt = wkt + SML;          // Wv^T [1024][4096]
    __bf16* wot = wvt + SML;          // Wo^T [4096][4096]
    __bf16* qb  = wot + BIG;          // Q    [B*S][4096]
    __bf16* kbf = qb  + BIG;          // K    [B][NKV][S][HD]
    __bf16* vtb = kbf + SML;          // V^T  [B][NKV][HD][S]
    __bf16* ab  = vtb + SML;          // attn [B*S][4096]

    // 1. convert x -> bf16
    cvt_f32_bf16<<<(B_ * S_ * D_ / 4 + 255) / 256, 256, 0, stream>>>(
        x, xb, B_ * S_ * D_);
    // 2. transpose-convert weights -> bf16 [N][K]
    transpose_cvt<<<dim3(4096 / 32, 4096 / 32), dim3(32, 8), 0, stream>>>(Wq, wqt, 4096, 4096);
    transpose_cvt<<<dim3(1024 / 32, 4096 / 32), dim3(32, 8), 0, stream>>>(Wk, wkt, 4096, 1024);
    transpose_cvt<<<dim3(1024 / 32, 4096 / 32), dim3(32, 8), 0, stream>>>(Wv, wvt, 4096, 1024);
    transpose_cvt<<<dim3(4096 / 32, 4096 / 32), dim3(32, 8), 0, stream>>>(Wo, wot, 4096, 4096);

    // 3. projections (M = B*S = 4096, 32-row M tiles)
    gemm_bf16<0><<<dim3(4096 / 256, 4096 / 32), 128, 0, stream>>>(
        xb, wqt, 4096, 4096, 4096, nullptr, qb);
    gemm_bf16<1><<<dim3(1024 / 256, 4096 / 32), 128, 0, stream>>>(
        xb, wkt, 4096, 1024, 4096, k_cache, kbf);
    gemm_bf16<2><<<dim3(1024 / 256, 4096 / 32), 128, 0, stream>>>(
        xb, wvt, 4096, 1024, 4096, v_cache, vtb);

    // 4. causal GQA flash attention
    gqa_flash<<<dim3(S_ / 16 / 4, B_ * NH), 128, 0, stream>>>(qb, kbf, vtb, ab);

    // 5. output projection -> fp32 out
    gemm_bf16<3><<<dim3(4096 / 256, 4096 / 32), 128, 0, stream>>>(
        ab, wot, 4096, 4096, 4096, out, nullptr);
}
